// QKVAttentionTwo_13245679141500
// MI455X (gfx1250) — matrix operand order repaired and checked by reading, not compiled, not run on hardware
//
#include <hip/hip_runtime.h>

// ---------------- problem constants ----------------
#define Bv   8
#define Sv   1024
#define Cv   512
#define Hv   8
#define DHv  64

typedef _Float16 v16h  __attribute__((ext_vector_type(16)));
typedef float    v8f   __attribute__((ext_vector_type(8)));
typedef unsigned int u32x4 __attribute__((ext_vector_type(4)));
typedef int      i32x8 __attribute__((ext_vector_type(8)));
typedef int      i32x4 __attribute__((ext_vector_type(4)));

union V16U { v16h v; uint4 q[2]; };          // fragment = two 16-byte runs

// A-fragment (16-bit A 16x32, ISA 7.12.2): lane = row, halves needed are the
// two contiguous runs [8g .. 8g+7] and [16+8g .. 16+8g+7] of the K-chunk.
// B-fragment: lane = column, halves are the same two runs [8g .. 8g+7] and [16+8g .. 16+8g+7].

// ---------------------------------------------------------------------
// TDM: async 2D f16 tile load Global -> LDS (CDNA5 ISA ch.8 D# layout).
// All sizes in f16 elements. Issued by one wave; completion = TENSORcnt.
// ---------------------------------------------------------------------
__device__ __forceinline__ void tdm_load_2d_f16(
    unsigned lds_off_bytes, const _Float16* gptr,
    unsigned tensor_d0, unsigned tensor_d1, unsigned d0_stride,
    unsigned tile_d0, unsigned tile_d1)
{
    unsigned long long ga = (unsigned long long)(size_t)gptr;
    u32x4 g0;
    g0[0] = 1u;                                       // count=1 (valid), user mode
    g0[1] = lds_off_bytes;                            // lds_addr
    g0[2] = (unsigned)(ga & 0xFFFFFFFFu);             // global_addr[31:0]
    g0[3] = (unsigned)((ga >> 32) & 0x1FFFFFFu)       // global_addr[56:32]
            | (2u << 30);                             // type = 2 ("image")
    i32x8 g1;
    g1[0] = (int)(1u << 16);                          // data_size=1 (2 bytes), mask=0
    g1[1] = (int)((tensor_d0 & 0xFFFFu) << 16);       // tensor_dim0[15:0]
    g1[2] = (int)(((tensor_d0 >> 16) & 0xFFFFu) |
                  ((tensor_d1 & 0xFFFFu) << 16));     // dim0 hi | dim1 lo
    g1[3] = (int)(((tensor_d1 >> 16) & 0xFFFFu) |
                  ((tile_d0 & 0xFFFFu) << 16));       // dim1 hi | tile_dim0
    g1[4] = (int)(tile_d1 & 0xFFFFu);                 // tile_dim1 (tile_dim2=0)
    g1[5] = (int)d0_stride;                           // tensor_dim0_stride[31:0]
    g1[6] = 0;
    g1[7] = 0;
    i32x4 z4 = {0, 0, 0, 0};
#if defined(__clang_major__) && (__clang_major__ >= 23)
    i32x8 z8 = {0, 0, 0, 0, 0, 0, 0, 0};
    __builtin_amdgcn_tensor_load_to_lds(g0, g1, z4, z4, z8, 0);
#else
    __builtin_amdgcn_tensor_load_to_lds(g0, g1, z4, z4, 0);
#endif
}

// =====================================================================
// Kernel 0: elementwise fp32 -> f16 (x and weights converted ONCE).
// =====================================================================
__global__ __launch_bounds__(256) void cvt_f32_f16(
    const float* __restrict__ in, _Float16* __restrict__ out, int n4)
{
    const int i = blockIdx.x * blockDim.x + threadIdx.x;
    if (i < n4) {
        float4 f = reinterpret_cast<const float4*>(in)[i];
        union { _Float16 h[4]; uint2 u; } o;
        o.h[0] = (_Float16)f.x; o.h[1] = (_Float16)f.y;
        o.h[2] = (_Float16)f.z; o.h[3] = (_Float16)f.w;
        reinterpret_cast<uint2*>(out)[i] = o.u;
    }
}

// =====================================================================
// Kernel 1: QKV projection (all-f16 operands).  y = x @ W^T + b.
// blockIdx.z: 0=Q (scaled 1/sqrt(DH)), 1=K, 2=V (stored transposed).
// 4 waves/block; each wave computes a 16x64 strip (A reused 4x).
// =====================================================================
__global__ __launch_bounds__(128) void proj_qkv(
    const _Float16* __restrict__ Xh,
    const _Float16* __restrict__ Wqh, const _Float16* __restrict__ Wkh,
    const _Float16* __restrict__ Wvh,
    const float* __restrict__ bq, const float* __restrict__ bk, const float* __restrict__ bv,
    _Float16* __restrict__ Qh, _Float16* __restrict__ Kh, _Float16* __restrict__ Vt)
{
    const int z = blockIdx.z;
    const _Float16* W    = (z == 0) ? Wqh : (z == 1) ? Wkh : Wvh;
    const float*    bias = (z == 0) ? bq  : (z == 1) ? bk  : bv;

    const int lane = threadIdx.x & 31;
    const int wave = threadIdx.x >> 5;
    const int l16  = lane & 15;
    const int g    = lane >> 4;

    const int m0 = blockIdx.y * 64 + wave * 16;   // row tile (b*S+s)
    const int n0 = blockIdx.x * 64;               // 64-wide channel strip

    v8f acc[4] = {};
    const _Float16* xrow = Xh + (size_t)(m0 + l16) * Cv;

    for (int k0 = 0; k0 < Cv; k0 += 32) {
        V16U a;
        a.q[0] = *reinterpret_cast<const uint4*>(xrow + k0 + 8 * g);
        a.q[1] = *reinterpret_cast<const uint4*>(xrow + k0 + 16 + 8 * g);
#pragma unroll
        for (int nt = 0; nt < 4; ++nt) {
            const _Float16* wrow = W + (size_t)(n0 + 16 * nt + l16) * Cv + k0 + 8 * g;
            V16U bm;
            bm.q[0] = *reinterpret_cast<const uint4*>(wrow);
            bm.q[1] = *reinterpret_cast<const uint4*>(wrow + 16);
            acc[nt] = __builtin_amdgcn_wmma_f32_16x16x32_f16(false, a.v, false, bm.v,
                                                             (short)0, acc[nt], false, false);
        }
    }

    const float scale = (z == 0) ? 0.125f : 1.0f;   // 1/sqrt(64) folded into Q
#pragma unroll
    for (int nt = 0; nt < 4; ++nt) {
        const int o = n0 + 16 * nt + l16;
        const int h = o >> 6;
        const int d = o & (DHv - 1);
        const float bval = bias[o];
#pragma unroll
        for (int v = 0; v < 8; ++v) {
            const int r  = m0 + 8 * g + v;     // flat row = b*S + s
            const int b_ = r >> 10;
            const int s  = r & (Sv - 1);
            const float val = (acc[nt][v] + bval) * scale;
            if (z == 2) {
                Vt[(((size_t)(b_ * Hv + h)) * DHv + d) * Sv + s] = (_Float16)val;
            } else {
                _Float16* O = (z == 0) ? Qh : Kh;
                O[(((size_t)(b_ * Hv + h)) * Sv + s) * DHv + d] = (_Float16)val;
            }
        }
    }
}

// =====================================================================
// Kernel 2: flash attention.  grid = (S/64, B*H), block = 128 (4 waves).
// Wave 0 TDM-loads the shared 32-key K tile (32x64) and V^T tile (64x32)
// into LDS; all 4 waves assemble WMMA fragments with ds_load_b128.
// =====================================================================
__global__ __launch_bounds__(128) void flash_attn(
    const _Float16* __restrict__ Qh, const _Float16* __restrict__ Kh,
    const _Float16* __restrict__ Vt, _Float16* __restrict__ Ctx)
{
    __shared__ __align__(16) _Float16 ldsK[32 * 64];   // [key][ch]
    __shared__ __align__(16) _Float16 ldsV[64 * 32];   // [ch][key]
    __shared__ __align__(16) _Float16 ldsP[4][16 * 32];

    const int lane = threadIdx.x & 31;
    const int wave = threadIdx.x >> 5;
    const int l16  = lane & 15;
    const int g    = lane >> 4;
    const int bh   = blockIdx.y;                    // b*H + h
    const int q0   = blockIdx.x * 64 + wave * 16;
    const int b_   = bh >> 3;
    const int h    = bh & 7;

    const _Float16* Qbase = Qh + ((size_t)bh * Sv + q0) * DHv;
    const _Float16* Kbase = Kh + (size_t)bh * Sv * DHv;
    const _Float16* Vbase = Vt + (size_t)bh * DHv * Sv;
    _Float16* Lp = ldsP[wave];
    const unsigned ldsK_off = (unsigned)(size_t)(void*)&ldsK[0];
    const unsigned ldsV_off = (unsigned)(size_t)(void*)&ldsV[0];

    // ---- Q A-fragments for DH chunks 0..31 and 32..63 (held in VGPRs) ----
    v16h qa[2];
#pragma unroll
    for (int c = 0; c < 2; ++c) {
        V16U u;
        u.q[0] = *reinterpret_cast<const uint4*>(Qbase + (size_t)l16 * DHv + 32 * c + 8 * g);
        u.q[1] = *reinterpret_cast<const uint4*>(Qbase + (size_t)l16 * DHv + 32 * c + 16 + 8 * g);
        qa[c] = u.v;
    }

    v8f acc[4] = {};
    float rm[8], rl[8];
#pragma unroll
    for (int v = 0; v < 8; ++v) { rm[v] = -1e30f; rl[v] = 0.0f; }

    for (int j0 = 0; j0 < Sv; j0 += 32) {
        __syncthreads();                 // previous tile fully consumed
        if (wave == 0) {
            // K tile: 32 keys x 64 ch from [S,DH];  V tile: 64 ch x 32 keys from [DH,S]
            tdm_load_2d_f16(ldsK_off, Kbase + (size_t)j0 * DHv,
                            DHv, Sv, DHv, DHv, 32);
            tdm_load_2d_f16(ldsV_off, Vbase + j0,
                            Sv, DHv, Sv, 32, DHv);
            if (j0 + 32 < Sv) {          // L2 prefetch of next tiles
                __builtin_prefetch(Kbase + (size_t)(j0 + 32 + l16) * DHv, 0, 1);
                __builtin_prefetch(Vbase + (size_t)l16 * Sv + j0 + 32, 0, 1);
            }
            __builtin_amdgcn_s_wait_tensorcnt(0);
        }
        __syncthreads();                 // tile visible to all waves

        // ---- scores: two 16-key sub-tiles, K-dim = DH via 2 WMMAs each ----
        v8f sc[2];
#pragma unroll
        for (int t = 0; t < 2; ++t) {
            const _Float16* kp = &ldsK[(16 * t + l16) * 64 + 8 * g];
            V16U kb0, kb1;
            kb0.q[0] = *reinterpret_cast<const uint4*>(kp);
            kb0.q[1] = *reinterpret_cast<const uint4*>(kp + 16);
            kb1.q[0] = *reinterpret_cast<const uint4*>(kp + 32);
            kb1.q[1] = *reinterpret_cast<const uint4*>(kp + 48);
            v8f cfrag = {};
            cfrag = __builtin_amdgcn_wmma_f32_16x16x32_f16(false, qa[0], false, kb0.v,
                                                           (short)0, cfrag, false, false);
            cfrag = __builtin_amdgcn_wmma_f32_16x16x32_f16(false, qa[1], false, kb1.v,
                                                           (short)0, cfrag, false, false);
            sc[t] = cfrag;
        }

        // ---- online softmax (row m = 8g+v lives across 16 lanes of group g) ----
        float p0[8], p1[8], av[8];
#pragma unroll
        for (int v = 0; v < 8; ++v) {
            float mx = fmaxf(sc[0][v], sc[1][v]);
            for (int off = 8; off; off >>= 1) mx = fmaxf(mx, __shfl_xor(mx, off, 16));
            const float nm = fmaxf(rm[v], mx);
            av[v] = __expf(rm[v] - nm);
            p0[v] = __expf(sc[0][v] - nm);
            p1[v] = __expf(sc[1][v] - nm);
            float rs = p0[v] + p1[v];
            for (int off = 8; off; off >>= 1) rs += __shfl_xor(rs, off, 16);
            rl[v] = rl[v] * av[v] + rs;
            rm[v] = nm;
        }
#pragma unroll
        for (int dt = 0; dt < 4; ++dt)
#pragma unroll
            for (int v = 0; v < 8; ++v) acc[dt][v] *= av[v];

        // ---- P: C-layout -> per-wave LDS -> A-layout (b128 reads) ----
#pragma unroll
        for (int v = 0; v < 8; ++v) {
            const int row = 8 * g + v;
            Lp[row * 32 + l16]      = (_Float16)p0[v];
            Lp[row * 32 + 16 + l16] = (_Float16)p1[v];
        }
        V16U pa;
        pa.q[0] = *reinterpret_cast<const uint4*>(&Lp[l16 * 32 + 8 * g]);
        pa.q[1] = *reinterpret_cast<const uint4*>(&Lp[l16 * 32 + 16 + 8 * g]);

        // ---- ctx += P(16x32) . V(32x16) for each of 4 channel tiles ----
#pragma unroll
        for (int dt = 0; dt < 4; ++dt) {
            const _Float16* vp = &ldsV[(16 * dt + l16) * 32 + 8 * g];
            V16U vb;
            vb.q[0] = *reinterpret_cast<const uint4*>(vp);
            vb.q[1] = *reinterpret_cast<const uint4*>(vp + 16);
            acc[dt] = __builtin_amdgcn_wmma_f32_16x16x32_f16(false, pa.v, false, vb.v,
                                                             (short)0, acc[dt], false, false);
        }
    }

    // ---- epilogue: divide by softmax denom, store ctx f16 [B,S,C] ----
#pragma unroll
    for (int dt = 0; dt < 4; ++dt)
#pragma unroll
        for (int v = 0; v < 8; ++v) {
            const int s  = q0 + 8 * g + v;
            const int ch = h * DHv + 16 * dt + l16;
            Ctx[((size_t)(b_ * Sv + s)) * Cv + ch] = (_Float16)(acc[dt][v] / rl[v]);
        }
}

// =====================================================================
// Kernel 3: output projection + bias + skip (all-f16 operands).
// =====================================================================
__global__ __launch_bounds__(128) void proj_out(
    const _Float16* __restrict__ Ctx, const _Float16* __restrict__ Woh,
    const float* __restrict__ bo, const float* __restrict__ Skip,
    float* __restrict__ Out)
{
    const int lane = threadIdx.x & 31;
    const int wave = threadIdx.x >> 5;
    const int l16  = lane & 15;
    const int g    = lane >> 4;
    const int m0   = blockIdx.y * 64 + wave * 16;
    const int n0   = blockIdx.x * 64;

    v8f acc[4] = {};
    const _Float16* crow = Ctx + (size_t)(m0 + l16) * Cv;

    for (int k0 = 0; k0 < Cv; k0 += 32) {
        V16U a;
        a.q[0] = *reinterpret_cast<const uint4*>(crow + k0 + 8 * g);
        a.q[1] = *reinterpret_cast<const uint4*>(crow + k0 + 16 + 8 * g);
#pragma unroll
        for (int nt = 0; nt < 4; ++nt) {
            const _Float16* wrow = Woh + (size_t)(n0 + 16 * nt + l16) * Cv + k0 + 8 * g;
            V16U bm;
            bm.q[0] = *reinterpret_cast<const uint4*>(wrow);
            bm.q[1] = *reinterpret_cast<const uint4*>(wrow + 16);
            acc[nt] = __builtin_amdgcn_wmma_f32_16x16x32_f16(false, a.v, false, bm.v,
                                                             (short)0, acc[nt], false, false);
        }
    }

#pragma unroll
    for (int nt = 0; nt < 4; ++nt) {
        const int   o    = n0 + 16 * nt + l16;
        const float bval = bo[o];
#pragma unroll
        for (int v = 0; v < 8; ++v) {
            const int r = m0 + 8 * g + v;
            Out[(size_t)r * Cv + o] = acc[nt][v] + bval + Skip[(size_t)r * Cv + o];
        }
    }
}

// =====================================================================
extern "C" void kernel_launch(void* const* d_in, const int* in_sizes, int n_in,
                              void* d_out, int out_size, void* d_ws, size_t ws_size,
                              hipStream_t stream) {
    (void)in_sizes; (void)n_in; (void)out_size; (void)ws_size;

    const float* X    = (const float*)d_in[0];
    const float* Skip = (const float*)d_in[1];
    const float* Wq   = (const float*)d_in[2];
    const float* bq   = (const float*)d_in[3];
    const float* Wk   = (const float*)d_in[4];
    const float* bk   = (const float*)d_in[5];
    const float* Wv   = (const float*)d_in[6];
    const float* bv   = (const float*)d_in[7];
    const float* Wo   = (const float*)d_in[8];
    const float* bo   = (const float*)d_in[9];
    float* Out        = (float*)d_out;

    const size_t NELEM = (size_t)Bv * Sv * Cv;          // 4,194,304
    const size_t WELEM = (size_t)Cv * Cv;               // 262,144
    char* ws = (char*)d_ws;
    _Float16* Qh   = (_Float16*)(ws);                   //  8 MB
    _Float16* Kh   = (_Float16*)(ws + 2 * NELEM);       //  8 MB
    _Float16* Vt   = (_Float16*)(ws + 4 * NELEM);       //  8 MB
    _Float16* Ctxh = (_Float16*)(ws + 6 * NELEM);       //  8 MB
    _Float16* Xh   = (_Float16*)(ws + 8 * NELEM);       //  8 MB
    _Float16* Wqh  = (_Float16*)(ws + 10 * NELEM);
    _Float16* Wkh  = (_Float16*)(ws + 10 * NELEM + 2 * WELEM);
    _Float16* Wvh  = (_Float16*)(ws + 10 * NELEM + 4 * WELEM);
    _Float16* Woh  = (_Float16*)(ws + 10 * NELEM + 6 * WELEM);

    // --- one-time fp32 -> f16 conversions ---
    cvt_f32_f16<<<dim3((unsigned)(NELEM / 4 + 255) / 256), dim3(256), 0, stream>>>(X,  Xh,  (int)(NELEM / 4));
    cvt_f32_f16<<<dim3((unsigned)(WELEM / 4 + 255) / 256), dim3(256), 0, stream>>>(Wq, Wqh, (int)(WELEM / 4));
    cvt_f32_f16<<<dim3((unsigned)(WELEM / 4 + 255) / 256), dim3(256), 0, stream>>>(Wk, Wkh, (int)(WELEM / 4));
    cvt_f32_f16<<<dim3((unsigned)(WELEM / 4 + 255) / 256), dim3(256), 0, stream>>>(Wv, Wvh, (int)(WELEM / 4));
    cvt_f32_f16<<<dim3((unsigned)(WELEM / 4 + 255) / 256), dim3(256), 0, stream>>>(Wo, Woh, (int)(WELEM / 4));

    dim3 blk(128);
    proj_qkv<<<dim3(Cv / 64, (Bv * Sv) / 64, 3), blk, 0, stream>>>(
        Xh, Wqh, Wkh, Wvh, bq, bk, bv, Qh, Kh, Vt);
    flash_attn<<<dim3(Sv / 64, Bv * Hv), blk, 0, stream>>>(Qh, Kh, Vt, Ctxh);
    proj_out<<<dim3(Cv / 64, (Bv * Sv) / 64), blk, 0, stream>>>(
        Ctxh, Woh, bo, Skip, Out);
}
